// SignMLPBert3_24979529794124
// MI455X (gfx1250) — matrix operand order repaired and checked by reading, not compiled, not run on hardware
//
#include <hip/hip_runtime.h>
#include <hip/hip_bf16.h>
#include <cstdint>
#include <cmath>

// ---------------------------------------------------------------------------
// SignMLPBert3 forward for gfx1250 (CDNA5, wave32, WMMA).
// Big GEMMs -> v_wmma_f32_16x16x32_f16, 128x64 block tiles, 32x32 per wave
// (8 WMMA per barrier), double-buffered LDS, b128 staging, pipelined loads.
// ---------------------------------------------------------------------------

typedef __attribute__((ext_vector_type(16))) _Float16 v16h;
typedef __attribute__((ext_vector_type(8)))  _Float16 v8h;
typedef __attribute__((ext_vector_type(8)))  float    v8f;

// ---------------------------- f32 -> f16 convert ---------------------------
__global__ void k_f32_to_f16(const float* __restrict__ s, _Float16* __restrict__ d, long n){
  long i = (long)blockIdx.x * blockDim.x + threadIdx.x;
  long st = (long)gridDim.x * blockDim.x;
  for (; i < n; i += st) d[i] = (_Float16)s[i];
}

// ------------------------------- WMMA GEMM ---------------------------------
// C[M,N] = A[M,K](f16, row stride lda) * W[K,N](f16) + bias
// EPI: 0 = none, 1 = leaky_relu(batchnorm(x)), 2 = relu
// Requires M%128==0, N%64==0, K%64==0.
template<int EPI, typename OUT_T>
__global__ __launch_bounds__(256)
void k_gemm_wmma(const _Float16* __restrict__ A, int lda,
                 const _Float16* __restrict__ W,
                 const float* __restrict__ bias,
                 const float* __restrict__ bn_g, const float* __restrict__ bn_b,
                 OUT_T* __restrict__ out, int M, int N, int K)
{
  constexpr int RS    = 72;            // LDS row stride in halves (16B-aligned frags)
  constexpr int AREG  = 128 * RS;      // A region (halves)
  constexpr int BREG  = 64 * RS;       // B region (halves)
  constexpr int BUFSZ = AREG + BREG;   // one pipeline stage
  __shared__ _Float16 lds[2 * BUFSZ];  // 55.3 KB double-buffered

  const int bm   = blockIdx.y * 128;
  const int bn   = blockIdx.x * 64;
  const int tid  = threadIdx.x;
  const int wave = tid >> 5;
  const int lane = tid & 31;
  const int wm   = wave >> 1;          // 0..3 : 32-row band
  const int wn   = wave & 1;           // 0..1 : 32-col band
  const int lh   = lane >> 4;          // lane half per ISA frag layout
  const int lm   = lane & 15;

  // global staging assignment (b128 per load)
  const int ar  = tid >> 1;            // 0..127 A row
  const int ac  = (tid & 1) << 5;      // 0/32   A col group (32 halves each)
  const int bk  = tid >> 3;            // 0..31  B k-row
  const int bn8 = (tid & 7) << 3;      // 0..56  B col group
  const _Float16* Ap = A + (size_t)(bm + ar) * lda + ac;
  const _Float16* Wp = W + (size_t)bk * N + bn + bn8;
  const size_t WpStep = (size_t)64 * N;
  const size_t WpHalf = (size_t)32 * N;

  v8f acc00 = {}, acc01 = {}, acc10 = {}, acc11 = {};
  const int ntiles = K >> 6;

  v8h a0, a1, a2, a3, b0, b1;

  // ---- prologue: stage tile 0 ----
  a0 = *(const v8h*)(Ap);      a1 = *(const v8h*)(Ap + 8);
  a2 = *(const v8h*)(Ap + 16); a3 = *(const v8h*)(Ap + 24);
  b0 = *(const v8h*)(Wp);      b1 = *(const v8h*)(Wp + WpHalf);
  Ap += 64; Wp += WpStep;
  {
    _Float16* As = lds;
    _Float16* Bs = lds + AREG;
    *(v8h*)&As[ar * RS + ac]      = a0;
    *(v8h*)&As[ar * RS + ac + 8]  = a1;
    *(v8h*)&As[ar * RS + ac + 16] = a2;
    *(v8h*)&As[ar * RS + ac + 24] = a3;
#pragma unroll
    for (int j = 0; j < 8; ++j) Bs[(bn8 + j) * RS + bk]      = b0[j];
#pragma unroll
    for (int j = 0; j < 8; ++j) Bs[(bn8 + j) * RS + bk + 32] = b1[j];
  }
  __syncthreads();

  for (int t = 0; t < ntiles; ++t) {
    const int cur = t & 1;
    _Float16* Asc = lds + cur * BUFSZ;
    _Float16* Bsc = Asc + AREG;
    const bool more = (t + 1) < ntiles;
    if (more) {                        // issue next-tile global loads early
      a0 = *(const v8h*)(Ap);      a1 = *(const v8h*)(Ap + 8);
      a2 = *(const v8h*)(Ap + 16); a3 = *(const v8h*)(Ap + 24);
      b0 = *(const v8h*)(Wp);      b1 = *(const v8h*)(Wp + WpHalf);
      __builtin_prefetch(Ap + 64, 0, 1);
      __builtin_prefetch(Wp + WpStep, 0, 1);
      Ap += 64; Wp += WpStep;
    }
    // compute: 2 k-slices x (2 row frags x 2 col frags) = 8 WMMA / barrier
#pragma unroll
    for (int kk = 0; kk < 64; kk += 32) {
      union { v16h v; v8h h[2]; } fa0, fa1, fb0, fb1;
      const int ab = kk + lh * 8;      // A: lane<16 K0..7/K16..23 ; lane>=16 +8
      fa0.h[0] = *(const v8h*)&Asc[(wm * 32 + lm) * RS + ab];
      fa0.h[1] = *(const v8h*)&Asc[(wm * 32 + lm) * RS + ab + 16];
      fa1.h[0] = *(const v8h*)&Asc[(wm * 32 + 16 + lm) * RS + ab];
      fa1.h[1] = *(const v8h*)&Asc[(wm * 32 + 16 + lm) * RS + ab + 16];
      const int bb = kk + lh * 8;      // B: lane<16 K0..7/K16..23 ; lane>=16 +8
      fb0.h[0] = *(const v8h*)&Bsc[(wn * 32 + lm) * RS + bb];
      fb0.h[1] = *(const v8h*)&Bsc[(wn * 32 + lm) * RS + bb + 16];
      fb1.h[0] = *(const v8h*)&Bsc[(wn * 32 + 16 + lm) * RS + bb];
      fb1.h[1] = *(const v8h*)&Bsc[(wn * 32 + 16 + lm) * RS + bb + 16];
      acc00 = __builtin_amdgcn_wmma_f32_16x16x32_f16(false, fa0.v, false, fb0.v,
                                                     (short)0, acc00, false, false);
      acc01 = __builtin_amdgcn_wmma_f32_16x16x32_f16(false, fa0.v, false, fb1.v,
                                                     (short)0, acc01, false, false);
      acc10 = __builtin_amdgcn_wmma_f32_16x16x32_f16(false, fa1.v, false, fb0.v,
                                                     (short)0, acc10, false, false);
      acc11 = __builtin_amdgcn_wmma_f32_16x16x32_f16(false, fa1.v, false, fb1.v,
                                                     (short)0, acc11, false, false);
    }
    if (more) {                        // stage next tile into alternate buffer
      _Float16* Asn = lds + (cur ^ 1) * BUFSZ;
      _Float16* Bsn = Asn + AREG;
      *(v8h*)&Asn[ar * RS + ac]      = a0;
      *(v8h*)&Asn[ar * RS + ac + 8]  = a1;
      *(v8h*)&Asn[ar * RS + ac + 16] = a2;
      *(v8h*)&Asn[ar * RS + ac + 24] = a3;
#pragma unroll
      for (int j = 0; j < 8; ++j) Bsn[(bn8 + j) * RS + bk]      = b0[j];
#pragma unroll
      for (int j = 0; j < 8; ++j) Bsn[(bn8 + j) * RS + bk + 32] = b1[j];
    }
    __syncthreads();
  }

  // --- epilogue + store (C/D layout: vgpr r -> M=r or r+8; N = lane%16) ---
  const int col0 = bn + wn * 32 + lm;
  const int col1 = col0 + 16;
  const float bv0 = bias ? bias[col0] : 0.0f;
  const float bv1 = bias ? bias[col1] : 0.0f;
  float bg0 = 0.f, bb0 = 0.f, bg1 = 0.f, bb1 = 0.f;
  if (EPI == 1) {
    bg0 = bn_g[col0] * 0.99999500003749969f; bb0 = bn_b[col0];
    bg1 = bn_g[col1] * 0.99999500003749969f; bb1 = bn_b[col1];
  }
#pragma unroll
  for (int r = 0; r < 8; ++r) {
    int row0 = bm + wm * 32 + r + lh * 8;
    int row1 = row0 + 16;
    float v00 = acc00[r] + bv0, v01 = acc01[r] + bv1;
    float v10 = acc10[r] + bv0, v11 = acc11[r] + bv1;
    if (EPI == 1) {
      v00 = v00 * bg0 + bb0; v00 = (v00 >= 0.f) ? v00 : 0.01f * v00;
      v01 = v01 * bg1 + bb1; v01 = (v01 >= 0.f) ? v01 : 0.01f * v01;
      v10 = v10 * bg0 + bb0; v10 = (v10 >= 0.f) ? v10 : 0.01f * v10;
      v11 = v11 * bg1 + bb1; v11 = (v11 >= 0.f) ? v11 : 0.01f * v11;
    }
    if (EPI == 2) {
      v00 = fmaxf(v00, 0.f); v01 = fmaxf(v01, 0.f);
      v10 = fmaxf(v10, 0.f); v11 = fmaxf(v11, 0.f);
    }
    out[(size_t)row0 * N + col0] = (OUT_T)v00;
    out[(size_t)row0 * N + col1] = (OUT_T)v01;
    out[(size_t)row1 * N + col0] = (OUT_T)v10;
    out[(size_t)row1 * N + col1] = (OUT_T)v11;
  }
}

// ------------------------- embedding + pos projection ----------------------
__global__ __launch_bounds__(256)
void k_embed(const float* __restrict__ x, const float* __restrict__ y, const float* __restrict__ z,
             const int* __restrict__ type_ids, const int* __restrict__ lm_ids,
             const float* __restrict__ type_embed, const float* __restrict__ type_g, const float* __restrict__ type_b,
             const float* __restrict__ lm_embed, const float* __restrict__ lm_g, const float* __restrict__ lm_b,
             const float* __restrict__ pos_w, const float* __restrict__ pos_b,
             _Float16* __restrict__ cat, int F, int L)
{
  const int t = blockIdx.x;
  const int c = threadIdx.x; // 0..255 == E
  __shared__ float r1[256], r2[256];

  // LN(type_embed[type]) , eps 1e-5
  int ty = type_ids[t];
  float e = type_embed[ty * 256 + c];
  r1[c] = e; r2[c] = e * e; __syncthreads();
  for (int off = 128; off > 0; off >>= 1) {
    if (c < off) { r1[c] += r1[c + off]; r2[c] += r2[c + off]; } __syncthreads();
  }
  float mean = r1[0] * (1.0f / 256.0f);
  float var  = r2[0] * (1.0f / 256.0f) - mean * mean;
  float inv  = rsqrtf(var + 1e-5f);
  cat[(size_t)t * 768 + c] = (_Float16)((e - mean) * inv * type_g[c] + type_b[c]);
  __syncthreads();

  // LN(lm_embed[landmark]) , eps 1e-5
  int lmk = lm_ids[t];
  float e2 = lm_embed[lmk * 256 + c];
  r1[c] = e2; r2[c] = e2 * e2; __syncthreads();
  for (int off = 128; off > 0; off >>= 1) {
    if (c < off) { r1[c] += r1[c + off]; r2[c] += r2[c + off]; } __syncthreads();
  }
  mean = r1[0] * (1.0f / 256.0f);
  var  = r2[0] * (1.0f / 256.0f) - mean * mean;
  inv  = rsqrtf(var + 1e-5f);
  cat[(size_t)t * 768 + 256 + c] = (_Float16)((e2 - mean) * inv * lm_g[c] + lm_b[c]);

  // pos9 @ pos_w + pos_b   (cur, next-frame, prev-frame xyz)
  int fl = t % (F * L);
  int f  = fl / L;
  float p9[9];
  p9[0] = x[t]; p9[1] = y[t]; p9[2] = z[t];
  if (f < F - 1) { p9[3] = x[t + L]; p9[4] = y[t + L]; p9[5] = z[t + L]; }
  else           { p9[3] = 0.f; p9[4] = 0.f; p9[5] = 0.f; }
  if (f > 0)     { p9[6] = x[t - L]; p9[7] = y[t - L]; p9[8] = z[t - L]; }
  else           { p9[6] = 0.f; p9[7] = 0.f; p9[8] = 0.f; }
  float accp = pos_b[c];
#pragma unroll
  for (int j = 0; j < 9; ++j) accp += p9[j] * pos_w[j * 256 + c];
  cat[(size_t)t * 768 + 512 + c] = (_Float16)accp;
}

// ------------------------------ row LayerNorm ------------------------------
__global__ __launch_bounds__(256)
void k_ln(const float* __restrict__ x, const float* __restrict__ res,
          const float* __restrict__ g, const float* __restrict__ b,
          float* __restrict__ out, int C, float eps)
{
  const int row = blockIdx.x;
  const int tid = threadIdx.x;
  __shared__ float r1[256], r2[256];
  float s = 0.f, s2 = 0.f;
  for (int c = tid; c < C; c += 256) {
    float v = x[(size_t)row * C + c];
    if (res) v += res[(size_t)row * C + c];
    s += v; s2 += v * v;
  }
  r1[tid] = s; r2[tid] = s2; __syncthreads();
  for (int off = 128; off > 0; off >>= 1) {
    if (tid < off) { r1[tid] += r1[tid + off]; r2[tid] += r2[tid + off]; } __syncthreads();
  }
  float mean = r1[0] / C;
  float var  = r2[0] / C - mean * mean;
  float inv  = rsqrtf(var + eps);
  for (int c = tid; c < C; c += 256) {
    float v = x[(size_t)row * C + c];
    if (res) v += res[(size_t)row * C + c];
    out[(size_t)row * C + c] = (v - mean) * inv * g[c] + b[c];
  }
}

// -------------------- disentangled attention (per b,head) ------------------
__device__ __forceinline__ float sgnf(int r){ return (r > 0) ? 1.f : ((r < 0) ? -1.f : 0.f); }
__device__ __forceinline__ int log_bucket(int rel){ // bucket=40, maxp=40, mid=20
  int a = rel < 0 ? -rel : rel;
  float absp = (a < 20) ? 19.0f : (float)a;
  float logp = ceilf(logf(absp / 20.0f) / logf(39.0f / 20.0f) * 19.0f) + 20.0f;
  float r = (absp <= 20.0f) ? (float)rel : logp * sgnf(rel);
  return (int)r;
}

__global__ __launch_bounds__(256)
void k_attn(const float* __restrict__ Q, const float* __restrict__ K, const float* __restrict__ V,
            const float* __restrict__ posq, const float* __restrict__ posk,
            const float* __restrict__ mask, float* __restrict__ ctx,
            int H, int d, float scale)
{
  const int S = 40;
  const int bh = blockIdx.x;
  const int b  = bh >> 4;
  const int h  = bh & 15;
  __shared__ float sc[40 * 40];
  __shared__ float pr[40 * 40];

  for (int idx = threadIdx.x; idx < S * S; idx += blockDim.x) {
    int q = idx / S, k = idx % S;
    const float* qv = Q + (size_t)(b * S + q) * H + h * d;
    const float* kv = K + (size_t)(b * S + k) * H + h * d;
    float s0 = 0.f;
    for (int dd = 0; dd < d; ++dd) s0 += qv[dd] * kv[dd];
    int c2p = log_bucket(q - k) + 40; c2p = min(max(c2p, 0), 79);
    int p2c = -log_bucket(k - q) + 40; p2c = min(max(p2c, 0), 79);
    const float* pk = posk + (size_t)c2p * H + h * d;
    const float* pq = posq + (size_t)p2c * H + h * d;
    float s1 = 0.f, s2 = 0.f;
    for (int dd = 0; dd < d; ++dd) { s1 += qv[dd] * pk[dd]; s2 += kv[dd] * pq[dd]; }
    sc[idx] = (s0 + s1 + s2) / scale;
  }
  __syncthreads();

  if (threadIdx.x < S) {
    int q = threadIdx.x;
    float mq = mask[b * S + q];
    float mx = -1e30f;
    for (int k = 0; k < S; ++k) {
      bool mm = mq * mask[b * S + k] > 0.f;
      float v = mm ? sc[q * S + k] : -1e9f;
      sc[q * S + k] = v; mx = fmaxf(mx, v);
    }
    float sum = 0.f;
    for (int k = 0; k < S; ++k) { float e = __expf(sc[q * S + k] - mx); pr[q * S + k] = e; sum += e; }
    float invs = 1.0f / sum;
    for (int k = 0; k < S; ++k) {
      bool mm = mq * mask[b * S + k] > 0.f;
      pr[q * S + k] = mm ? pr[q * S + k] * invs : 0.f;
    }
  }
  __syncthreads();

  for (int idx = threadIdx.x; idx < S * d; idx += blockDim.x) {
    int q = idx / d, dd = idx % d;
    float acc = 0.f;
    for (int k = 0; k < S; ++k) acc += pr[q * S + k] * V[(size_t)(b * S + k) * H + h * d + dd];
    ctx[(size_t)(b * S + q) * H + h * d + dd] = acc;
  }
}

// ------------------------------- small kernels -----------------------------
__global__ void k_small_gemm(const float* __restrict__ A, const float* __restrict__ W,
                             const float* __restrict__ bias, float* __restrict__ out,
                             int M, int N, int Kd)
{
  int idx = blockIdx.x * blockDim.x + threadIdx.x;
  if (idx >= M * N) return;
  int r = idx / N, c = idx % N;
  float acc = bias ? bias[c] : 0.f;
  for (int k = 0; k < Kd; ++k) acc += A[(size_t)r * Kd + k] * W[(size_t)k * N + c];
  out[idx] = acc;
}

__global__ void k_maskmul(const float* __restrict__ in, const float* __restrict__ mask,
                          float* __restrict__ out, int C)
{
  int row = blockIdx.x;
  float m = mask[row];
  for (int c = threadIdx.x; c < C; c += blockDim.x)
    out[(size_t)row * C + c] = in[(size_t)row * C + c] * m;
}

__global__ void k_add_f16(const float* __restrict__ a, const float* __restrict__ b,
                          _Float16* __restrict__ o, long n)
{
  long i = (long)blockIdx.x * blockDim.x + threadIdx.x;
  long st = (long)gridDim.x * blockDim.x;
  for (; i < n; i += st) o[i] = (_Float16)(a[i] + b[i]);
}

__global__ __launch_bounds__(512)
void k_fuse(const float* __restrict__ full, const float* __restrict__ lh, const float* __restrict__ rh,
            const float* __restrict__ lips, const float* __restrict__ face, _Float16* __restrict__ fused)
{
  int row = blockIdx.x; int c = threadIdx.x;
  size_t r = (size_t)row * 512 + c;
  size_t o = (size_t)row * 2048;
  fused[o + c]        = (_Float16)full[r];
  fused[o + 512 + c]  = (_Float16)fmaxf(lh[r], rh[r]);
  fused[o + 1024 + c] = (_Float16)lips[r];
  fused[o + 1536 + c] = (_Float16)face[r];
}

__global__ void k_pool(const float* __restrict__ h, const float* __restrict__ mask,
                       float* __restrict__ pooled, int C)
{
  int idx = blockIdx.x * blockDim.x + threadIdx.x;
  if (idx >= 32 * C) return;
  int b = idx / C, c = idx % C;
  float s = 0.f, ms = 0.f;
  for (int t = 0; t < 40; ++t) {
    float m = mask[b * 40 + t];
    s += h[(size_t)(b * 40 + t) * C + c] * m;
    ms += m;
  }
  pooled[idx] = s / ms;
}

// ---------------------------------------------------------------------------
extern "C" void kernel_launch(void* const* d_in, const int* in_sizes, int n_in,
                              void* d_out, int out_size, void* d_ws, size_t ws_size,
                              hipStream_t stream)
{
  (void)in_sizes; (void)n_in; (void)out_size; (void)ws_size;
  const int B = 32, F = 40, L = 100, E = 256, TOK = B * F * L, BF = B * F;

  // -------- input tensor map (setup_inputs dict order, nested dicts flattened) --------
  const float* in_x   = (const float*)d_in[0];
  const float* in_y   = (const float*)d_in[1];
  const float* in_z   = (const float*)d_in[2];
  const int*   in_ty  = (const int*)d_in[3];
  const int*   in_lm  = (const int*)d_in[4];
  const float* in_mask= (const float*)d_in[5];
  const float* type_embed=(const float*)d_in[6];  const float* lm_embed=(const float*)d_in[7];
  const float* type_g=(const float*)d_in[8];      const float* type_b=(const float*)d_in[9];
  const float* lm_g=(const float*)d_in[10];       const float* lm_b=(const float*)d_in[11];
  const float* pos_w=(const float*)d_in[12];      const float* pos_b=(const float*)d_in[13];
  const float* dense_w=(const float*)d_in[14];    const float* dense_b=(const float*)d_in[15];
  const float* logits_w=(const float*)d_in[16];   const float* logits_b=(const float*)d_in[17];
  const float* br_w[6];  const float* br_b[6];  const float* br_g[6];  const float* br_be[6];
  for (int i = 0; i < 6; ++i) {
    br_w[i]  = (const float*)d_in[18 + 4 * i + 0];
    br_b[i]  = (const float*)d_in[18 + 4 * i + 1];
    br_g[i]  = (const float*)d_in[18 + 4 * i + 2];
    br_be[i] = (const float*)d_in[18 + 4 * i + 3];
  }
  struct EncW {
    const float *qw,*qb,*kw,*kb,*vw,*vb,*sow,*sob,*so_g,*so_b,*iw,*ib,*ow,*ob,*out_g,*out_b,*rel,*rel_g,*rel_b;
    int H, O;
  } enc[3];
  const int bases[3] = {42, 61, 80};
  const int Hs[3] = {384, 512, 640};
  const int Os[3] = {512, 640, 640};
  for (int i = 0; i < 3; ++i) {
    int bs = bases[i];
    enc[i].qw=(const float*)d_in[bs+0];  enc[i].qb=(const float*)d_in[bs+1];
    enc[i].kw=(const float*)d_in[bs+2];  enc[i].kb=(const float*)d_in[bs+3];
    enc[i].vw=(const float*)d_in[bs+4];  enc[i].vb=(const float*)d_in[bs+5];
    enc[i].sow=(const float*)d_in[bs+6]; enc[i].sob=(const float*)d_in[bs+7];
    enc[i].so_g=(const float*)d_in[bs+8];enc[i].so_b=(const float*)d_in[bs+9];
    enc[i].iw=(const float*)d_in[bs+10]; enc[i].ib=(const float*)d_in[bs+11];
    enc[i].ow=(const float*)d_in[bs+12]; enc[i].ob=(const float*)d_in[bs+13];
    enc[i].out_g=(const float*)d_in[bs+14]; enc[i].out_b=(const float*)d_in[bs+15];
    enc[i].rel=(const float*)d_in[bs+16];  enc[i].rel_g=(const float*)d_in[bs+17];
    enc[i].rel_b=(const float*)d_in[bs+18];
    enc[i].H = Hs[i]; enc[i].O = Os[i];
  }

  // -------- workspace bump allocator --------
  char* wsb = (char*)d_ws;
  size_t off = 0;
  auto alloc = [&](size_t bytes) -> void* {
    size_t a = (off + 255) & ~(size_t)255;
    off = a + bytes;
    return (void*)(wsb + a);
  };

  _Float16* cat768 = (_Float16*)alloc((size_t)TOK * 768 * 2);
  _Float16* fts16  = (_Float16*)alloc((size_t)TOK * 256 * 2);
  _Float16* dense_w16 = (_Float16*)alloc((size_t)768 * 256 * 2);
  const int brK[6] = {21*E, 21*E, 21*E, 25*E, 100*E, 2048};
  const int brN[6] = {512, 512, 512, 512, 512, 384};
  _Float16* br_w16[6];
  for (int i = 0; i < 6; ++i) br_w16[i] = (_Float16*)alloc((size_t)brK[i] * brN[i] * 2);
  float* bout[5];
  for (int i = 0; i < 5; ++i) bout[i] = (float*)alloc((size_t)BF * 512 * 4);
  _Float16* fused16 = (_Float16*)alloc((size_t)BF * 2048 * 2);
  float* lmkout = (float*)alloc((size_t)BF * 384 * 4);
  float* hcur  = (float*)alloc((size_t)BF * 640 * 4);
  float* hnext = (float*)alloc((size_t)BF * 640 * 4);
  _Float16* h16 = (_Float16*)alloc((size_t)BF * 640 * 2);
  _Float16* a16 = (_Float16*)alloc((size_t)BF * 640 * 2);
  float* Qf = (float*)alloc((size_t)BF * 640 * 4);
  float* Kf = (float*)alloc((size_t)BF * 640 * 4);
  float* Vf = (float*)alloc((size_t)BF * 640 * 4);
  float* ctx = (float*)alloc((size_t)BF * 640 * 4);
  float* attnout = (float*)alloc((size_t)BF * 640 * 4);
  float* inter = (float*)alloc((size_t)BF * 640 * 4);
  float* tmp = (float*)alloc((size_t)BF * 640 * 4);
  float* relln = (float*)alloc((size_t)80 * 640 * 4);
  float* posq = (float*)alloc((size_t)80 * 640 * 4);
  float* posk = (float*)alloc((size_t)80 * 640 * 4);
  float* pooled = (float*)alloc((size_t)32 * 640 * 4);
  _Float16 *eqw16[3], *ekw16[3], *evw16[3], *esow16[3], *eiw16[3], *eow16[3];
  for (int i = 0; i < 3; ++i) {
    size_t hh = (size_t)Hs[i] * Hs[i] * 2, ho = (size_t)Hs[i] * Os[i] * 2;
    eqw16[i]=(_Float16*)alloc(hh); ekw16[i]=(_Float16*)alloc(hh); evw16[i]=(_Float16*)alloc(hh);
    esow16[i]=(_Float16*)alloc(hh); eiw16[i]=(_Float16*)alloc(hh); eow16[i]=(_Float16*)alloc(ho);
  }

  auto conv = [&](const float* s, _Float16* d, long n) {
    k_f32_to_f16<<<2048, 256, 0, stream>>>(s, d, n);
  };

  // -------- weight conversion to f16 --------
  conv(dense_w, dense_w16, (long)768 * 256);
  for (int i = 0; i < 6; ++i) conv(br_w[i], br_w16[i], (long)brK[i] * brN[i]);
  for (int i = 0; i < 3; ++i) {
    long hh = (long)Hs[i] * Hs[i], ho = (long)Hs[i] * Os[i];
    conv(enc[i].qw, eqw16[i], hh);  conv(enc[i].kw, ekw16[i], hh);
    conv(enc[i].vw, evw16[i], hh);  conv(enc[i].sow, esow16[i], hh);
    conv(enc[i].iw, eiw16[i], hh);  conv(enc[i].ow, eow16[i], ho);
  }

  // -------- token features: embeddings + pos projection -> cat768 (f16) ----
  k_embed<<<TOK, 256, 0, stream>>>(in_x, in_y, in_z, in_ty, in_lm,
                                   type_embed, type_g, type_b,
                                   lm_embed, lm_g, lm_b,
                                   pos_w, pos_b, cat768, F, L);

  // fts = cat768 @ dense_w + dense_b   [128000 x 256], f16 out
  k_gemm_wmma<0, _Float16><<<dim3(256/64, TOK/128), 256, 0, stream>>>(
      cat768, 768, dense_w16, dense_b, nullptr, nullptr, fts16, TOK, 256, 768);

  // -------- branch MLPs (landmark index sets are contiguous ranges) --------
  // lh: lm 0..20, rh: 21..41, lips: 42..62, face: 63..87, full: 0..99
  const int brStart[5] = {0, 21, 42, 63, 0};
  const int brKi[5]    = {21*E, 21*E, 21*E, 25*E, 100*E};
  for (int i = 0; i < 5; ++i) {
    k_gemm_wmma<1, float><<<dim3(512/64, BF/128), 256, 0, stream>>>(
        fts16 + (size_t)brStart[i] * E, 100 * E, br_w16[i], br_b[i],
        br_g[i], br_be[i], bout[i], BF, 512, brKi[i]);
  }
  // fused = [full, max(lh,rh), lips, face] -> f16 [1280 x 2048]
  k_fuse<<<BF, 512, 0, stream>>>(bout[4], bout[0], bout[1], bout[2], bout[3], fused16);
  // lmk branch -> [1280 x 384]
  k_gemm_wmma<1, float><<<dim3(384/64, BF/128), 256, 0, stream>>>(
      fused16, 2048, br_w16[5], br_b[5], br_g[5], br_be[5], lmkout, BF, 384, 2048);
  // fts = fts * m1
  k_maskmul<<<BF, 256, 0, stream>>>(lmkout, in_mask, hcur, 384);

  // ------------------------------ 3 DeBERTa layers -------------------------
  for (int l = 0; l < 3; ++l) {
    const int H = enc[l].H, O = enc[l].O, d = H / 16;
    const float scale = sqrtf((float)d * 3.0f);

    conv(hcur, h16, (long)BF * H);
    // Q, K, V
    k_gemm_wmma<0, float><<<dim3(H/64, BF/128), 256, 0, stream>>>(
        h16, H, eqw16[l], enc[l].qb, nullptr, nullptr, Qf, BF, H, H);
    k_gemm_wmma<0, float><<<dim3(H/64, BF/128), 256, 0, stream>>>(
        h16, H, ekw16[l], enc[l].kb, nullptr, nullptr, Kf, BF, H, H);
    k_gemm_wmma<0, float><<<dim3(H/64, BF/128), 256, 0, stream>>>(
        h16, H, evw16[l], enc[l].vb, nullptr, nullptr, Vf, BF, H, H);
    // rel embedding LN (eps 1e-7) + pos projections (tiny: plain kernels)
    k_ln<<<80, 256, 0, stream>>>(enc[l].rel, nullptr, enc[l].rel_g, enc[l].rel_b, relln, H, 1e-7f);
    int pn = 80 * H;
    k_small_gemm<<<(pn + 255) / 256, 256, 0, stream>>>(relln, enc[l].qw, enc[l].qb, posq, 80, H, H);
    k_small_gemm<<<(pn + 255) / 256, 256, 0, stream>>>(relln, enc[l].kw, enc[l].kb, posk, 80, H, H);
    // attention with c2p/p2c disentangled bias
    k_attn<<<B * 16, 256, 0, stream>>>(Qf, Kf, Vf, posq, posk, in_mask, ctx, H, d, scale);
    // self-output + residual LN
    conv(ctx, a16, (long)BF * H);
    k_gemm_wmma<0, float><<<dim3(H/64, BF/128), 256, 0, stream>>>(
        a16, H, esow16[l], enc[l].sob, nullptr, nullptr, tmp, BF, H, H);
    k_ln<<<BF, 256, 0, stream>>>(tmp, hcur, enc[l].so_g, enc[l].so_b, attnout, H, 1e-7f);
    // intermediate (ReLU fused)
    conv(attnout, a16, (long)BF * H);
    k_gemm_wmma<2, float><<<dim3(H/64, BF/128), 256, 0, stream>>>(
        a16, H, eiw16[l], enc[l].ib, nullptr, nullptr, inter, BF, H, H);
    // (inter + attn_out) @ ow + ob -> LN
    k_add_f16<<<2048, 256, 0, stream>>>(inter, attnout, a16, (long)BF * H);
    k_gemm_wmma<0, float><<<dim3(O/64, BF/128), 256, 0, stream>>>(
        a16, H, eow16[l], enc[l].ob, nullptr, nullptr, tmp, BF, O, H);
    k_ln<<<BF, 256, 0, stream>>>(tmp, nullptr, enc[l].out_g, enc[l].out_b, hnext, O, 1e-7f);
    // swap
    float* t2 = hcur; hcur = hnext; hnext = t2;
  }

  // ------------------------------ pool + logits ----------------------------
  k_pool<<<(32 * 640 + 255) / 256, 256, 0, stream>>>(hcur, in_mask, pooled, 640);
  k_small_gemm<<<(32 * 250 + 255) / 256, 256, 0, stream>>>(
      pooled, logits_w, logits_b, (float*)d_out, 32, 250, 640);
}
